// GAT_65549790871805
// MI455X (gfx1250) — matrix operand order repaired and checked by reading, not compiled, not run on hardware
//
#include <hip/hip_runtime.h>
#include <hip/hip_bf16.h>
#include <math.h>

typedef __attribute__((ext_vector_type(16))) _Float16 v16h;
typedef __attribute__((ext_vector_type(8)))  float    v8f;
typedef __attribute__((ext_vector_type(4)))  float    v4f;

#define NEG_SLOPE 0.2f

// -------- ordered-uint encoding for float atomic max (exact, deterministic) -----
__device__ __forceinline__ unsigned fkey(float f) {
  unsigned u = __float_as_uint(f);
  return (u & 0x80000000u) ? ~u : (u | 0x80000000u);
}

// -------- pack weight matrix [K][NC] fp32 -> WMMA f16 B fragments ----------------
// Fragment layout (16-bit B, 32x16, wave32): col = lane&15;
// lanes 0-15: elems 0..7 = K+0..7, elems 8..15 = K+16..23; lanes 16-31: +8.
// Packed as [ks][tile][lane] x 16 halves, contiguous 32B per lane.
__global__ void pack_w_kernel(const float* __restrict__ W, _Float16* __restrict__ Bp,
                              int K, int NC, int numTiles) {
  int idx = blockIdx.x * blockDim.x + threadIdx.x;
  int total = (K >> 5) * numTiles * 32;
  if (idx >= total) return;
  int lane  = idx & 31;
  int t     = (idx >> 5) % numTiles;
  int ks    = idx / (32 * numTiles);
  int col   = t * 16 + (lane & 15);
  int kbase = ks * 32 + ((lane >= 16) ? 8 : 0);
  _Float16* out = Bp + (size_t)idx * 16;
#pragma unroll
  for (int i = 0; i < 16; ++i) {
    int kk  = kbase + i + ((i >= 8) ? 8 : 0);
    float v = (kk < K && col < NC) ? W[(size_t)kk * NC + col] : 0.0f;
    out[i] = (_Float16)v;
  }
}

// -------- WMMA GEMM: C[Nrows][NC] = A[Nrows][K] (fp32, converted to f16) @ Bp ----
// One wave computes a 16-row x (NT*16)-col slab. K multiple of 32.
// A fragment (16-bit A, 16x32): row = m0 + (lane&15);
// lanes 0-15: elems 0..7 = K+0..7, elems 8..15 = K+16..23; lanes 16-31: +8.
// STREAM_A=true emits non-temporal loads for A (read-once stream; preserves the
// 192MB L2 for the edge-phase working set on MI455X).
template<int NT, bool STREAM_A>
__global__ void wmma_gemm_kernel(const float* __restrict__ A, const _Float16* __restrict__ Bp,
                                 float* __restrict__ C, int Nrows, int K, int NC) {
  int gwave = (int)((blockIdx.x * blockDim.x + threadIdx.x) >> 5);
  int lane  = threadIdx.x & 31;
  int m0    = gwave * 16;
  if (m0 >= Nrows) return;                    // wave-uniform: EXEC stays all-ones
  int hi  = (lane >= 16) ? 1 : 0;
  int row = m0 + (lane & 15);
  if (row >= Nrows) row = Nrows - 1;          // clamp (N is a multiple of 16 here)
  const v4f* ap = (const v4f*)(A + (size_t)row * K + hi * 8);

  v8f acc[NT] = {};
  int ksteps = K >> 5;
  for (int ks = 0; ks < ksteps; ++ks) {
    const v4f* q = ap + (size_t)ks * 8;       // ks*32 floats
    v4f f0, f1, f2, f3;
    if (STREAM_A) {
      f0 = __builtin_nontemporal_load(q + 0);
      f1 = __builtin_nontemporal_load(q + 1);
      f2 = __builtin_nontemporal_load(q + 4);
      f3 = __builtin_nontemporal_load(q + 5);
    } else {
      f0 = q[0]; f1 = q[1]; f2 = q[4]; f3 = q[5];
    }
    v16h a;
    a[0]=(_Float16)f0.x; a[1]=(_Float16)f0.y; a[2]=(_Float16)f0.z; a[3]=(_Float16)f0.w;
    a[4]=(_Float16)f1.x; a[5]=(_Float16)f1.y; a[6]=(_Float16)f1.z; a[7]=(_Float16)f1.w;
    a[8]=(_Float16)f2.x; a[9]=(_Float16)f2.y; a[10]=(_Float16)f2.z; a[11]=(_Float16)f2.w;
    a[12]=(_Float16)f3.x; a[13]=(_Float16)f3.y; a[14]=(_Float16)f3.z; a[15]=(_Float16)f3.w;
#pragma unroll
    for (int t = 0; t < NT; ++t) {
      v16h b = *((const v16h*)Bp + ((size_t)ks * NT + t) * 32 + lane);
      acc[t] = __builtin_amdgcn_wmma_f32_16x16x32_f16(
          false, a, false, b, (short)0, acc[t], false, false);
    }
  }
  // C/D layout: VGPR r -> row m0 + r + hi*8, col = tile*16 + (lane&15)
#pragma unroll
  for (int t = 0; t < NT; ++t) {
    int col = t * 16 + (lane & 15);
    if (col < NC) {
#pragma unroll
      for (int r = 0; r < 8; ++r) {
        int ro = m0 + r + hi * 8;
        if (ro < Nrows) C[(size_t)ro * NC + col] = acc[t][r];
      }
    }
  }
}

// -------- per-node attention coefficients el/er ----------------------------------
__global__ void el_er_kernel(const float* __restrict__ h, const float* __restrict__ al,
                             const float* __restrict__ ar, float* __restrict__ el,
                             float* __restrict__ er, int N, int H, int F) {
  int idx = blockIdx.x * blockDim.x + threadIdx.x;  // idx = n*H + head
  if (idx >= N * H) return;
  int hd = idx % H;
  const float* hp = h + (size_t)idx * F;            // [N][H*F] contiguous
  float sl = 0.f, sr = 0.f;
  for (int f = 0; f < F; ++f) {
    float v = hp[f];
    sl += v * al[hd * F + f];
    sr += v * ar[hd * F + f];
  }
  el[idx] = sl;
  er[idx] = sr;
}

// -------- edge pass 1: segment max via ordered-uint atomicMax --------------------
__global__ void edge_max_kernel(const int* __restrict__ src, const int* __restrict__ dst,
                                const float* __restrict__ el, const float* __restrict__ er,
                                unsigned* __restrict__ mkey, int E, int H) {
  int idx = blockIdx.x * blockDim.x + threadIdx.x;
  if (idx >= E * H) return;
  int e = idx / H, hd = idx - e * H;
  int s = src[e], d = dst[e];
  float v = el[(size_t)s * H + hd] + er[(size_t)d * H + hd];
  v = v > 0.f ? v : NEG_SLOPE * v;
  atomicMax(mkey + (size_t)d * H + hd, fkey(v));
}

__global__ void decode_max_kernel(unsigned* __restrict__ m, int total) {
  int i = blockIdx.x * blockDim.x + threadIdx.x;
  if (i >= total) return;
  unsigned k = m[i];
  float f;
  if (k == 0u) f = -INFINITY;  // node with no incoming edges
  else f = __uint_as_float((k & 0x80000000u) ? (k & 0x7FFFFFFFu) : ~k);
  ((float*)m)[i] = f;
}

// -------- edge pass 2: ex = exp(e - m[dst]); stash ex per edge; s[dst] += ex -----
__global__ void edge_sum_kernel(const int* __restrict__ src, const int* __restrict__ dst,
                                const float* __restrict__ el, const float* __restrict__ er,
                                const float* __restrict__ m, float* __restrict__ ssum,
                                float* __restrict__ exbuf, int E, int H) {
  int idx = blockIdx.x * blockDim.x + threadIdx.x;
  if (idx >= E * H) return;
  int e = idx / H, hd = idx - e * H;
  int s = src[e], d = dst[e];
  float v = el[(size_t)s * H + hd] + er[(size_t)d * H + hd];
  v = v > 0.f ? v : NEG_SLOPE * v;
  float ex = __expf(v - m[(size_t)d * H + hd]);
  exbuf[idx] = ex;
  atomicAdd(ssum + (size_t)d * H + hd, ex);
}

// -------- edge pass 3: rst[dst] += feat[src] * (ex / s[dst]) ---------------------
__global__ void edge_aggr_kernel(const int* __restrict__ src, const int* __restrict__ dst,
                                 const float* __restrict__ exbuf, const float* __restrict__ ssum,
                                 const float* __restrict__ feat, float* __restrict__ rst,
                                 int E, int H, int F) {
  int HF  = H * F;
  int idx = blockIdx.x * blockDim.x + threadIdx.x;  // idx = edge*HF + j
  if (idx >= E * HF) return;
  int e = idx / HF, j = idx - e * HF;
  int hd = j / F;
  int s = src[e], d = dst[e];
  float alpha = exbuf[(size_t)e * H + hd] / ssum[(size_t)d * H + hd];
  atomicAdd(rst + (size_t)d * HF + j, feat[(size_t)s * HF + j] * alpha);
}

// -------- layer-1 epilogue: bias + ELU -------------------------------------------
__global__ void bias_elu_kernel(const float* __restrict__ rst, const float* __restrict__ b,
                                float* __restrict__ out, int N, int HF) {
  int idx = blockIdx.x * blockDim.x + threadIdx.x;
  if (idx >= N * HF) return;
  float v = rst[idx] + b[idx % HF];
  out[idx] = v > 0.f ? v : (__expf(v) - 1.0f);
}

// -------- layer-2 epilogue: bias + log_softmax over 40 logits --------------------
__global__ void bias_logsoftmax40_kernel(const float* __restrict__ rst,
                                         const float* __restrict__ b,
                                         float* __restrict__ out, int N) {
  int n = blockIdx.x * blockDim.x + threadIdx.x;
  if (n >= N) return;
  float v[40];
  float mx = -INFINITY;
#pragma unroll
  for (int f = 0; f < 40; ++f) {
    v[f] = rst[(size_t)n * 40 + f] + b[f];
    mx = fmaxf(mx, v[f]);
  }
  float s = 0.f;
#pragma unroll
  for (int f = 0; f < 40; ++f) s += __expf(v[f] - mx);
  float lse = mx + __logf(s);
#pragma unroll
  for (int f = 0; f < 40; ++f) out[(size_t)n * 40 + f] = v[f] - lse;
}

extern "C" void kernel_launch(void* const* d_in, const int* in_sizes, int n_in,
                              void* d_out, int out_size, void* d_ws, size_t ws_size,
                              hipStream_t stream) {
  const float* x   = (const float*)d_in[0];
  const int*   src = (const int*)d_in[1];
  const int*   dst = (const int*)d_in[2];
  const float* W1  = (const float*)d_in[3];
  const float* al1 = (const float*)d_in[4];
  const float* ar1 = (const float*)d_in[5];
  const float* b1  = (const float*)d_in[6];
  const float* W2  = (const float*)d_in[7];
  const float* al2 = (const float*)d_in[8];
  const float* ar2 = (const float*)d_in[9];
  const float* b2  = (const float*)d_in[10];

  const int N = 100000, E = 1600000, K1 = 512;
  const int H1 = 8, F1 = 8, HF1 = 64;   // layer1: 8 heads x 8
  const int H2 = 1, F2 = 40, HF2 = 40;  // layer2: 1 head x 40
  (void)in_sizes; (void)n_in; (void)out_size; (void)ws_size; (void)F2;

  // ---- carve workspace (256B aligned chunks) ----
  char* w = (char*)d_ws;
  auto carve = [&](size_t bytes) -> char* {
    char* p = w;
    w += (bytes + 255) & ~(size_t)255;
    return p;
  };
  _Float16* W1p  = (_Float16*)carve((size_t)(K1 / 32) * 4 * 32 * 16 * sizeof(_Float16));
  _Float16* W2p  = (_Float16*)carve((size_t)(64 / 32) * 3 * 32 * 16 * sizeof(_Float16));
  float* h1pre   = (float*)carve((size_t)N * HF1 * 4);
  float* el1     = (float*)carve((size_t)N * H1 * 4);
  float* er1     = (float*)carve((size_t)N * H1 * 4);
  float* h1f     = (float*)carve((size_t)N * HF1 * 4);
  float* h2pre   = (float*)carve((size_t)N * HF2 * 4);
  float* el2     = (float*)carve((size_t)N * 4);
  float* er2     = (float*)carve((size_t)N * 4);
  float* ex1     = (float*)carve((size_t)E * H1 * 4);  // per-edge exp(e-m), layer1
  float* ex2     = (float*)carve((size_t)E * 4);       // per-edge exp(e-m), layer2
  // zero-initialized region (single memset)
  char* z0 = w;
  unsigned* m1   = (unsigned*)carve((size_t)N * H1 * 4);
  float* s1      = (float*)carve((size_t)N * H1 * 4);
  float* rst1    = (float*)carve((size_t)N * HF1 * 4);
  unsigned* m2   = (unsigned*)carve((size_t)N * 4);
  float* s2      = (float*)carve((size_t)N * 4);
  float* rst2    = (float*)carve((size_t)N * HF2 * 4);
  hipMemsetAsync(z0, 0, (size_t)(w - z0), stream);

  const int B = 256;
  // weight fragment packing
  pack_w_kernel<<<((K1 / 32) * 4 * 32 + B - 1) / B, B, 0, stream>>>(W1, W1p, K1, 64, 4);
  pack_w_kernel<<<1, B, 0, stream>>>(W2, W2p, 64, 40, 3);

  int waves   = (N + 15) / 16;         // 6250
  int gblocks = (waves + 7) / 8;       // 8 waves per 256-thread block

  // ---- layer 1 ----
  wmma_gemm_kernel<4, true><<<gblocks, B, 0, stream>>>(x, W1p, h1pre, N, K1, HF1);
  el_er_kernel<<<(N * H1 + B - 1) / B, B, 0, stream>>>(h1pre, al1, ar1, el1, er1, N, H1, F1);
  edge_max_kernel<<<(E * H1 + B - 1) / B, B, 0, stream>>>(src, dst, el1, er1, m1, E, H1);
  decode_max_kernel<<<(N * H1 + B - 1) / B, B, 0, stream>>>(m1, N * H1);
  edge_sum_kernel<<<(E * H1 + B - 1) / B, B, 0, stream>>>(src, dst, el1, er1, (float*)m1,
                                                          s1, ex1, E, H1);
  edge_aggr_kernel<<<(E * HF1 + B - 1) / B, B, 0, stream>>>(src, dst, ex1, s1,
                                                            h1pre, rst1, E, H1, F1);
  bias_elu_kernel<<<(N * HF1 + B - 1) / B, B, 0, stream>>>(rst1, b1, h1f, N, HF1);

  // ---- layer 2 ----
  wmma_gemm_kernel<3, false><<<gblocks, B, 0, stream>>>(h1f, W2p, h2pre, N, 64, HF2);
  el_er_kernel<<<(N * H2 + B - 1) / B, B, 0, stream>>>(h2pre, al2, ar2, el2, er2, N, H2, F2);
  edge_max_kernel<<<(E * H2 + B - 1) / B, B, 0, stream>>>(src, dst, el2, er2, m2, E, H2);
  decode_max_kernel<<<(N + B - 1) / B, B, 0, stream>>>(m2, N);
  edge_sum_kernel<<<(E * H2 + B - 1) / B, B, 0, stream>>>(src, dst, el2, er2, (float*)m2,
                                                          s2, ex2, E, H2);
  edge_aggr_kernel<<<(E * HF2 + B - 1) / B, B, 0, stream>>>(src, dst, ex2, s2,
                                                            h2pre, rst2, E, H2, F2);
  bias_logsoftmax40_kernel<<<(N + B - 1) / B, B, 0, stream>>>(rst2, b2, (float*)d_out, N);
}
